// _MambaBlockWrapper_15075335209602
// MI455X (gfx1250) — matrix operand order repaired and checked by reading, not compiled, not run on hardware
//
#include <hip/hip_runtime.h>
#include <hip/hip_bf16.h>
#include <cstdint>
#include <cstddef>

// ---------------- problem constants ----------------
#define B_    4
#define L_    2048
#define D_    1024
#define EV_   64
#define DI_   2048          // d_inner
#define NST   16            // d_state
#define KCONV 4
#define DTR   64            // dt_rank
#define ROWS  (B_ * L_)     // 8192
#define KAUG  (D_ + EV_)    // 1088
#define LNEPS 1e-5f

typedef __attribute__((ext_vector_type(16))) _Float16 v16h;
typedef __attribute__((ext_vector_type(8)))  float    v8f;
typedef __attribute__((ext_vector_type(4)))  unsigned int u32x4;

union FragU { v16h v; u32x4 q[2]; };

// epilogue selectors
#define EP_BIAS     0
#define EP_SOFTPLUS 1
#define EP_RESADD   2

// LDS byte offset of a __shared__ object: AS3->generic cast keeps the LDS
// offset in addr[31:0] (aperture lives in the high 32 bits per ISA 10.2).
__device__ __forceinline__ uint32_t lds_off(const void* p) {
    return (uint32_t)(uintptr_t)p;
}

// CDNA5 async global->LDS copy, 16 bytes per lane. Tracked by ASYNCcnt.
__device__ __forceinline__ void async_cp16(uint32_t lds_dst, const void* gsrc) {
    asm volatile("global_load_async_to_lds_b128 %0, %1, off"
                 :: "v"(lds_dst), "v"(gsrc)
                 : "memory");
}
__device__ __forceinline__ void wait_asynccnt0() {
    asm volatile("s_wait_asynccnt 0x0" ::: "memory");
}

// =====================================================================
// Tiled WMMA GEMM:  C[M,N] = A[M,K](f16, row-major) * Bt[N,K](f16, K-major)
// Block tile 128(M) x 64(N), K-step 32. 8 waves: 4(M) x 2(N), each wave
// owns a 32x32 output tile = 2x2 v_wmma_f32_16x16x32_f16 accumulators.
// LDS double-buffered; tiles staged with global_load_async_to_lds_b128
// (ASYNCcnt) while WMMAs consume the previous tile; one barrier per K-step.
// =====================================================================
template <int EPI>
__global__ __launch_bounds__(256) void gemm_wmma(
    const _Float16* __restrict__ A,    // [M,K] row-major f16
    const _Float16* __restrict__ Bt,   // [N,K] (i.e. B transposed) f16
    float* __restrict__ Cout,          // [M,N] f32
    const float* __restrict__ bias,    // [N] or nullptr
    const float* __restrict__ addsrc,  // [M,N] or nullptr (residual)
    int M, int N, int K)
{
    __shared__ _Float16 lA[2][128][48];   // padded rows: 96B stride, 16B aligned
    __shared__ _Float16 lB[2][64][48];

    const int tid  = threadIdx.x;
    const int wave = tid >> 5;
    const int lane = tid & 31;
    const int r16  = lane & 15;
    const int hid  = lane >> 4;       // half-wave id
    const int wm   = wave >> 1;       // 0..3
    const int wn   = wave & 1;        // 0..1

    const int m0 = blockIdx.y * 128;
    const int n0 = blockIdx.x * 64;

    // A stage mapping: 2 threads per row, 16 halves (2x16B) each
    const int ra = tid >> 1;          // 0..127
    const int sa = (tid & 1) * 16;    // half offset 0 / 16
    // B stage mapping: 4 threads per row, 8 halves (16B) each
    const int rb = tid >> 2;          // 0..63
    const int sb = (tid & 3) * 8;
    const bool bvalid = (n0 + rb) < N;

    // per-thread LDS destinations (both buffers)
    const uint32_t dstA0 = lds_off(&lA[0][ra][sa]);
    const uint32_t dstA1 = lds_off(&lA[1][ra][sa]);
    const uint32_t dstB0 = lds_off(&lB[0][rb][sb]);
    const uint32_t dstB1 = lds_off(&lB[1][rb][sb]);

    // rows past N are never staged: zero them once in both buffers
    if (!bvalid) {
        const u32x4 z = {};
        *(u32x4*)&lB[0][rb][sb] = z;
        *(u32x4*)&lB[1][rb][sb] = z;
    }

    const _Float16* gA = A  + (size_t)(m0 + ra) * K + sa;
    const _Float16* gB = Bt + (size_t)(n0 + (bvalid ? rb : 0)) * K + sb;

    auto stage = [&](int buf, int k0) {
        const uint32_t da = buf ? dstA1 : dstA0;
        async_cp16(da,      gA + k0);
        async_cp16(da + 16, gA + k0 + 8);
        if (bvalid) async_cp16(buf ? dstB1 : dstB0, gB + k0);
    };

    const v8f vzero = {};
    v8f acc[2][2];
    acc[0][0] = vzero; acc[0][1] = vzero; acc[1][0] = vzero; acc[1][1] = vzero;

    const int nk = K / 32;
    stage(0, 0);   // prologue

    for (int i = 0; i < nk; ++i) {
        const int cur = i & 1;
        wait_asynccnt0();     // our async writes into buffer `cur` are done
        __syncthreads();      // everyone's are visible

        if (i + 1 < nk) stage(cur ^ 1, (i + 1) * 32);   // overlap next tile

        const _Float16 (*lAc)[48] = lA[cur];
        const _Float16 (*lBc)[48] = lB[cur];

        FragU af[2], bf[2];
#pragma unroll
        for (int mt = 0; mt < 2; ++mt) {
            const _Float16* ap = &lAc[wm * 32 + mt * 16 + r16][0];
            af[mt].q[0] = *(const u32x4*)(ap + hid * 8);        // K 0..7 / 8..15
            af[mt].q[1] = *(const u32x4*)(ap + 16 + hid * 8);   // K 16..23 / 24..31
        }
#pragma unroll
        for (int nt = 0; nt < 2; ++nt) {
            const _Float16* bp = &lBc[wn * 32 + nt * 16 + r16][0];
            bf[nt].q[0] = *(const u32x4*)(bp + hid * 8);        // K 0..7 / 8..15
            bf[nt].q[1] = *(const u32x4*)(bp + 16 + hid * 8);   // K 16..23 / 24..31
        }
#pragma unroll
        for (int mt = 0; mt < 2; ++mt)
#pragma unroll
            for (int nt = 0; nt < 2; ++nt)
                acc[mt][nt] = __builtin_amdgcn_wmma_f32_16x16x32_f16(
                    false, af[mt].v, false, bf[nt].v,
                    (short)0, acc[mt][nt], false, false);
        // no trailing barrier: next iteration's wait+barrier protects reuse
    }

    // ---- epilogue: VGPR v -> row = base + 8*hid + v, col = base + r16 ----
#pragma unroll
    for (int mt = 0; mt < 2; ++mt) {
#pragma unroll
        for (int nt = 0; nt < 2; ++nt) {
            const int col = n0 + wn * 32 + nt * 16 + r16;
            if (col >= N) continue;
            const int rbase = m0 + wm * 32 + mt * 16 + hid * 8;
            const float bval = (bias != nullptr) ? bias[col] : 0.0f;
#pragma unroll
            for (int v = 0; v < 8; ++v) {
                float val = acc[mt][nt][v] + bval;
                if (EPI == EP_SOFTPLUS) {
                    val = (val > 20.0f) ? val : log1pf(__expf(val));
                } else if (EPI == EP_RESADD) {
                    val += addsrc[(size_t)(rbase + v) * N + col];
                }
                Cout[(size_t)(rbase + v) * N + col] = val;
            }
        }
    }
}

// =====================================================================
// Weight transpose + f32->f16:  in[K,N] row-major -> out[N,K] (K-major)
// =====================================================================
__global__ __launch_bounds__(256) void wconv_kernel(
    const float* __restrict__ in, _Float16* __restrict__ out, int K, int N)
{
    size_t idx = (size_t)blockIdx.x * 256 + threadIdx.x;
    size_t tot = (size_t)K * N;
    if (idx >= tot) return;
    int k = (int)(idx / N);
    int n = (int)(idx % N);
    out[(size_t)n * K + k] = (_Float16)in[idx];
}

// =====================================================================
// Pack [x | event_emb] -> f16 activation [ROWS, KAUG]
// =====================================================================
__global__ __launch_bounds__(256) void pack_act_kernel(
    const float* __restrict__ x, const float* __restrict__ ev,
    _Float16* __restrict__ out)
{
    size_t idx = (size_t)blockIdx.x * 256 + threadIdx.x;
    if (idx >= (size_t)ROWS * KAUG) return;
    int row = (int)(idx / KAUG);
    int c   = (int)(idx % KAUG);
    float v = (c < D_) ? x[(size_t)row * D_ + c]
                       : ev[(size_t)row * EV_ + (c - D_)];
    out[idx] = (_Float16)v;
}

// =====================================================================
// LayerNorm over D=1024; one 256-thread block per row; f16 output
// =====================================================================
__global__ __launch_bounds__(256) void ln_kernel(
    const float* __restrict__ aug, const float* __restrict__ g,
    const float* __restrict__ b, _Float16* __restrict__ hout)
{
    __shared__ float s1[256];
    __shared__ float s2[256];
    const int row = blockIdx.x;
    const int tid = threadIdx.x;
    const float* rp = aug + (size_t)row * D_;

    float sum = 0.f, sq = 0.f;
    float vloc[4];
#pragma unroll
    for (int i = 0; i < 4; ++i) {
        float v = rp[tid + i * 256];
        vloc[i] = v;
        sum += v;
        sq  += v * v;
    }
    s1[tid] = sum; s2[tid] = sq;
    __syncthreads();
    for (int off = 128; off > 0; off >>= 1) {
        if (tid < off) { s1[tid] += s1[tid + off]; s2[tid] += s2[tid + off]; }
        __syncthreads();
    }
    const float mu  = s1[0] * (1.0f / D_);
    const float var = s2[0] * (1.0f / D_) - mu * mu;
    const float rs  = rsqrtf(var + LNEPS);
#pragma unroll
    for (int i = 0; i < 4; ++i) {
        int c = tid + i * 256;
        float hv = (vloc[i] - mu) * rs * g[c] + b[c];
        hout[(size_t)row * D_ + c] = (_Float16)hv;
    }
}

// =====================================================================
// Causal depthwise conv1d (width 4) + SiLU; reads xh = xz[:, 0:DI]
// writes f32 (scan input) and f16 (x_proj GEMM input)
// =====================================================================
__global__ __launch_bounds__(256) void conv_silu_kernel(
    const float* __restrict__ xz, const float* __restrict__ cw,
    const float* __restrict__ cb, float* __restrict__ xcf,
    _Float16* __restrict__ xch)
{
    size_t idx = (size_t)blockIdx.x * 256 + threadIdx.x;
    if (idx >= (size_t)ROWS * DI_) return;
    int row = (int)(idx / DI_);
    int d   = (int)(idx % DI_);
    int l   = row % L_;
    float acc = cb[d];
#pragma unroll
    for (int k = 0; k < KCONV; ++k) {
        int ls = l - (KCONV - 1) + k;
        if (ls >= 0)
            acc += cw[d * KCONV + k] *
                   xz[(size_t)(row - (KCONV - 1) + k) * (2 * DI_) + d];
    }
    float s = acc / (1.0f + __expf(-acc));   // silu
    xcf[idx] = s;
    xch[idx] = (_Float16)s;
}

// =====================================================================
// Extract dt (first DTR cols of dbl) -> f16 GEMM input
// =====================================================================
__global__ __launch_bounds__(256) void extract_dt_kernel(
    const float* __restrict__ dbl, _Float16* __restrict__ dtA)
{
    size_t idx = (size_t)blockIdx.x * 256 + threadIdx.x;
    if (idx >= (size_t)ROWS * DTR) return;
    int row = (int)(idx / DTR);
    int j   = (int)(idx % DTR);
    dtA[idx] = (_Float16)dbl[(size_t)row * (DTR + 2 * NST) + j];
}

// =====================================================================
// Selective scan: 1 thread / channel, 16-state recurrence in registers.
// B/C (shared per timestep) double-buffered through LDS: one barrier
// per timestep (writes for step l go to buf l&1 before the barrier,
// step l+1 writes the other buffer -> no WAR across a single barrier).
// Fuses y = (scan + D*u) * silu(z) -> f16 for out_proj GEMM.
// =====================================================================
__global__ __launch_bounds__(64) void scan_kernel(
    const float* __restrict__ dtv,   // [ROWS, DI] softplus'd dt
    const float* __restrict__ xcf,   // [ROWS, DI] u
    const float* __restrict__ dbl,   // [ROWS, 96] (dt | B | C)
    const float* __restrict__ xz,    // [ROWS, 2*DI] (z at +DI)
    const float* __restrict__ A_log, // [DI, NST]
    const float* __restrict__ Dp,    // [DI]
    _Float16* __restrict__ yz)       // [ROWS, DI] f16 output
{
    const int b   = blockIdx.x;
    const int d   = blockIdx.y * 64 + threadIdx.x;
    __shared__ float sBC[2][2 * NST];   // [parity][B(16) | C(16)]

    float Av[NST];
#pragma unroll
    for (int n = 0; n < NST; ++n) Av[n] = -__expf(A_log[(size_t)d * NST + n]);
    const float Dd = Dp[d];

    float hs[NST];
#pragma unroll
    for (int n = 0; n < NST; ++n) hs[n] = 0.0f;

    for (int l = 0; l < L_; ++l) {
        const size_t row = (size_t)b * L_ + l;
        const int p = l & 1;
        if (threadIdx.x < 2 * NST)
            sBC[p][threadIdx.x] = dbl[row * (DTR + 2 * NST) + DTR + threadIdx.x];
        __syncthreads();

        const float dt = dtv[row * DI_ + d];
        const float u  = xcf[row * DI_ + d];
        const float z  = xz[row * (2 * DI_) + DI_ + d];
        const float du = dt * u;
        float y = 0.0f;
#pragma unroll
        for (int n = 0; n < NST; ++n) {
            hs[n] = __expf(dt * Av[n]) * hs[n] + du * sBC[p][n];
            y += hs[n] * sBC[p][NST + n];
        }
        float out = (y + Dd * u) * (z / (1.0f + __expf(-z)));
        yz[row * DI_ + d] = (_Float16)out;
        // no second barrier: next step writes the other parity buffer
    }
}

// =====================================================================
// Host-side orchestration
// =====================================================================
extern "C" void kernel_launch(void* const* d_in, const int* in_sizes, int n_in,
                              void* d_out, int out_size, void* d_ws, size_t ws_size,
                              hipStream_t stream)
{
    const float* x          = (const float*)d_in[0];
    const float* ev         = (const float*)d_in[1];
    const float* W_in       = (const float*)d_in[2];
    const float* b_in       = (const float*)d_in[3];
    const float* ln_g       = (const float*)d_in[4];
    const float* ln_b       = (const float*)d_in[5];
    const float* in_proj_w  = (const float*)d_in[6];
    const float* conv_w     = (const float*)d_in[7];
    const float* conv_b     = (const float*)d_in[8];
    const float* x_proj_w   = (const float*)d_in[9];
    const float* dt_proj_w  = (const float*)d_in[10];
    const float* dt_proj_b  = (const float*)d_in[11];
    const float* A_log      = (const float*)d_in[12];
    const float* D_param    = (const float*)d_in[13];
    const float* out_proj_w = (const float*)d_in[14];
    float* out = (float*)d_out;

    // ---- carve workspace (all 256B aligned) ----
    char* p = (char*)d_ws;
    auto carve = [&](size_t bytes) -> char* {
        char* r = p;
        p += (bytes + 255) & ~(size_t)255;
        return r;
    };
    _Float16* wT_in   = (_Float16*)carve((size_t)D_      * KAUG * 2); // [1024][1088]
    _Float16* wT_inp  = (_Float16*)carve((size_t)(2*DI_) * D_   * 2); // [4096][1024]
    _Float16* wT_xp   = (_Float16*)carve((size_t)(DTR+2*NST) * DI_ * 2); // [96][2048]
    _Float16* wT_dtp  = (_Float16*)carve((size_t)DI_     * DTR  * 2); // [2048][64]
    _Float16* wT_outp = (_Float16*)carve((size_t)D_      * DI_  * 2); // [1024][2048]
    _Float16* actA    = (_Float16*)carve((size_t)ROWS * KAUG * 2);
    float*    aug     = (float*)   carve((size_t)ROWS * D_   * 4);
    _Float16* hb      = (_Float16*)carve((size_t)ROWS * D_   * 2);
    float*    xzb     = (float*)   carve((size_t)ROWS * 2*DI_* 4);
    float*    xcf     = (float*)   carve((size_t)ROWS * DI_  * 4);
    _Float16* xch     = (_Float16*)carve((size_t)ROWS * DI_  * 2);
    float*    dbl     = (float*)   carve((size_t)ROWS * (DTR+2*NST) * 4);
    _Float16* dtA     = (_Float16*)carve((size_t)ROWS * DTR  * 2);
    float*    dtv     = (float*)   carve((size_t)ROWS * DI_  * 4);
    _Float16* yzb     = (_Float16*)carve((size_t)ROWS * DI_  * 2);
    (void)ws_size; (void)in_sizes; (void)n_in; (void)out_size;

    const dim3 blk(256);
    auto nblk = [](size_t tot) { return (unsigned)((tot + 255) / 256); };
    auto ggrid = [](int N) { return dim3((unsigned)((N + 63) / 64), ROWS / 128); };

    // 1) weight transpose+convert (f32 [K,N] -> f16 [N,K])
    wconv_kernel<<<nblk((size_t)KAUG * D_),      blk, 0, stream>>>(W_in,       wT_in,   KAUG, D_);
    wconv_kernel<<<nblk((size_t)D_ * 2*DI_),     blk, 0, stream>>>(in_proj_w,  wT_inp,  D_,   2*DI_);
    wconv_kernel<<<nblk((size_t)DI_*(DTR+2*NST)),blk, 0, stream>>>(x_proj_w,   wT_xp,   DI_,  DTR+2*NST);
    wconv_kernel<<<nblk((size_t)DTR * DI_),      blk, 0, stream>>>(dt_proj_w,  wT_dtp,  DTR,  DI_);
    wconv_kernel<<<nblk((size_t)DI_ * D_),       blk, 0, stream>>>(out_proj_w, wT_outp, DI_,  D_);

    // 2) pack [x | event_emb] -> f16
    pack_act_kernel<<<nblk((size_t)ROWS * KAUG), blk, 0, stream>>>(x, ev, actA);

    // 3) aug = actA @ W_in + b_in   (residual, f32)
    gemm_wmma<EP_BIAS><<<ggrid(D_), blk, 0, stream>>>(
        actA, wT_in, aug, b_in, nullptr, ROWS, D_, KAUG);

    // 4) layernorm -> f16
    ln_kernel<<<ROWS, blk, 0, stream>>>(aug, ln_g, ln_b, hb);

    // 5) xz = h @ in_proj_w  (f32, [ROWS, 4096])
    gemm_wmma<EP_BIAS><<<ggrid(2*DI_), blk, 0, stream>>>(
        hb, wT_inp, xzb, nullptr, nullptr, ROWS, 2*DI_, D_);

    // 6) causal depthwise conv + silu
    conv_silu_kernel<<<nblk((size_t)ROWS * DI_), blk, 0, stream>>>(
        xzb, conv_w, conv_b, xcf, xch);

    // 7) dbl = xc @ x_proj_w  (f32, [ROWS, 96])
    gemm_wmma<EP_BIAS><<<ggrid(DTR + 2*NST), blk, 0, stream>>>(
        xch, wT_xp, dbl, nullptr, nullptr, ROWS, DTR + 2*NST, DI_);

    // 8) extract dt columns -> f16
    extract_dt_kernel<<<nblk((size_t)ROWS * DTR), blk, 0, stream>>>(dbl, dtA);

    // 9) dtv = softplus(dtA @ dt_proj_w + dt_proj_b)
    gemm_wmma<EP_SOFTPLUS><<<ggrid(DI_), blk, 0, stream>>>(
        dtA, wT_dtp, dtv, dt_proj_b, nullptr, ROWS, DI_, DTR);

    // 10) selective scan -> yz f16 (fused + D*u and *silu(z))
    scan_kernel<<<dim3(B_, DI_ / 64), dim3(64), 0, stream>>>(
        dtv, xcf, dbl, xzb, A_log, D_param, yzb);

    // 11) out = aug + yz @ out_proj_w
    gemm_wmma<EP_RESADD><<<ggrid(D_), blk, 0, stream>>>(
        yzb, wT_outp, out, nullptr, aug, ROWS, D_, DI_);
}
